// ManualRNN_57836029608478
// MI455X (gfx1250) — matrix operand order repaired and checked by reading, not compiled, not run on hardware
//
#include <hip/hip_runtime.h>
#include <hip/hip_bf16.h>
#include <math.h>

#define SEQ  2048
#define BB   64
#define II   512
#define HH   512
#define MTOT (SEQ * BB)
#define NBLK 8   // scan blocks, one per 64-column slice of H

typedef __attribute__((ext_vector_type(16))) _Float16 v16h;
typedef __attribute__((ext_vector_type(8)))  _Float16 v8h;
typedef __attribute__((ext_vector_type(4)))  _Float16 v4h;
typedef __attribute__((ext_vector_type(8)))  float    v8f;
typedef __attribute__((ext_vector_type(4)))  unsigned v4u;
typedef __attribute__((ext_vector_type(8)))  int      v8i;
typedef __attribute__((ext_vector_type(4)))  int      v4i;

// ---- WMMA helpers ---------------------------------------------------------
__device__ __forceinline__ v8f wmma_f16(v16h a, v16h b, v8f c) {
    return __builtin_amdgcn_wmma_f32_16x16x32_f16(
        false, a, false, b, (short)0, c, false, false);
}

// A fragment (16x32 f16): lane m holds row m, K=[0..7]+[16..23]; lane m+16: K=[8..15]+[24..31]
__device__ __forceinline__ v16h frag_a(const _Float16* sm, int row_base, int ld,
                                       int k0, int lane) {
    const _Float16* p = sm + (row_base + (lane & 15)) * ld + k0 + ((lane & 16) ? 8 : 0);
    v8h lo = *(const v8h*)(p);
    v8h hi = *(const v8h*)(p + 16);
    return __builtin_shufflevector(lo, hi, 0,1,2,3,4,5,6,7,8,9,10,11,12,13,14,15);
}

// B fragment (32x16 f16): lane n holds column n (= W row n), K=[0..7]+[16..23]; lane n+16: K=[8..15]+[24..31]
__device__ __forceinline__ v16h frag_b(const _Float16* sm, int col_base, int ld,
                                       int k0, int lane) {
    const _Float16* p = sm + (col_base + (lane & 15)) * ld + k0 + ((lane & 16) ? 8 : 0);
    v8h lo = *(const v8h*)(p);
    v8h hi = *(const v8h*)(p + 16);
    return __builtin_shufflevector(lo, hi, 0,1,2,3,4,5,6,7,8,9,10,11,12,13,14,15);
}

// ---- Phase 1: x_proj = input @ W_ih^T + bias -> d_out ---------------------
// Block tile 128(M) x 256(N), K staged in 32-chunks, double buffered.
// 8 waves = 2(M) x 4(N); wave tile 64x64 = 16 accumulators, 16 WMMAs / k-tile.
#define LDA 40   // 32 + 8 halves padding (80B stride, 16B aligned, conflict-free)

__global__ __launch_bounds__(256) void xproj_kernel(
    const float* __restrict__ A,      // [MTOT, II]
    const float* __restrict__ W,      // [HH, II]
    const float* __restrict__ bih,
    const float* __restrict__ bhh,
    float* __restrict__ out)          // [MTOT, HH]
{
    __shared__ __align__(16) _Float16 As[2][128 * LDA];
    __shared__ __align__(16) _Float16 Bs[2][256 * LDA];

    const int tid  = threadIdx.x;
    const int lane = tid & 31;
    const int w    = tid >> 5;
    const int m0   = blockIdx.x * 128;
    const int n0   = blockIdx.y * 256;
    const int wm   = (w & 1) * 64;
    const int wn   = (w >> 1) * 64;

    v8f acc[4][4];
#pragma unroll
    for (int mi = 0; mi < 4; ++mi)
#pragma unroll
        for (int ni = 0; ni < 4; ++ni) acc[mi][ni] = (v8f){};

    auto loadA = [&](int kt, int buf) {
        const int k0 = kt * 32;
#pragma unroll
        for (int i = 0; i < 4; ++i) {
            int id = tid + 256 * i;          // 1024 float4 per tile
            int r  = id >> 3;
            int c4 = (id & 7) << 2;
            const float* gp = A + (size_t)(m0 + r) * II + k0 + c4;
            float4 v = *(const float4*)gp;
            __builtin_prefetch(gp + 64, 0, 1);   // speculative, k-tile+2
            v4h hv;
            hv[0] = (_Float16)v.x; hv[1] = (_Float16)v.y;
            hv[2] = (_Float16)v.z; hv[3] = (_Float16)v.w;
            *(v4h*)&As[buf][r * LDA + c4] = hv;
        }
    };
    auto loadB = [&](int kt, int buf) {
        const int k0 = kt * 32;
#pragma unroll
        for (int i = 0; i < 8; ++i) {
            int id = tid + 256 * i;          // 2048 float4 per tile
            int r  = id >> 3;
            int c4 = (id & 7) << 2;
            float4 v = *(const float4*)(W + (size_t)(n0 + r) * II + k0 + c4);
            v4h hv;
            hv[0] = (_Float16)v.x; hv[1] = (_Float16)v.y;
            hv[2] = (_Float16)v.z; hv[3] = (_Float16)v.w;
            *(v4h*)&Bs[buf][r * LDA + c4] = hv;
        }
    };

    loadA(0, 0);
    loadB(0, 0);
    __syncthreads();

    for (int kt = 0; kt < II / 32; ++kt) {
        const int buf = kt & 1;
        if (kt < II / 32 - 1) { loadA(kt + 1, buf ^ 1); loadB(kt + 1, buf ^ 1); }
        v16h af[4], bf[4];
#pragma unroll
        for (int mi = 0; mi < 4; ++mi) af[mi] = frag_a(As[buf], wm + mi * 16, LDA, 0, lane);
#pragma unroll
        for (int ni = 0; ni < 4; ++ni) bf[ni] = frag_b(Bs[buf], wn + ni * 16, LDA, 0, lane);
#pragma unroll
        for (int mi = 0; mi < 4; ++mi)
#pragma unroll
            for (int ni = 0; ni < 4; ++ni)
                acc[mi][ni] = wmma_f16(af[mi], bf[ni], acc[mi][ni]);
        __syncthreads();
    }

    // Epilogue: D layout — VGPR v: lanes<16 row v, lanes>=16 row v+8; col = lane&15
    const int rsel = (lane & 16) ? 8 : 0;
    const int cl   = lane & 15;
#pragma unroll
    for (int ni = 0; ni < 4; ++ni) {
        const int   col  = n0 + wn + ni * 16 + cl;
        const float bias = bih[col] + bhh[col];
#pragma unroll
        for (int mi = 0; mi < 4; ++mi) {
            const int rbase = m0 + wm + mi * 16 + rsel;
#pragma unroll
            for (int v = 0; v < 8; ++v) {
                out[(size_t)(rbase + v) * HH + col] = acc[mi][ni][v] + bias;
            }
        }
    }
}

// ---- Phase 2: persistent recurrent scan with TDM h-staging ----------------
// 8 blocks; block j owns H columns [64j,64j+64). W_hh slice (f16) resident in
// LDS. h is exchanged between blocks as f16 via a double-buffered scratch in
// d_ws, staged into LDS each step by TENSOR_LOAD_TO_LDS (TDM), with TDM pad
// recreating the 520-half padded LDS stride.
#define LDW 520  // 512 + 8 halves (1040B stride; (1040/4) mod 64 = 4 -> conflict-free)

__global__ __launch_bounds__(256) void rnn_scan_kernel(
    const float* __restrict__ Whh,    // [HH, HH]
    float* __restrict__ out,          // holds xp, overwritten with h (f32 output)
    unsigned int* __restrict__ ctr,   // [SEQ] zeroed step counters
    _Float16* __restrict__ h16)       // [2][BB*HH] f16 h exchange buffers
{
    __shared__ __align__(16) _Float16 Wsm[64 * LDW];
    __shared__ __align__(16) _Float16 Hsm[BB * LDW];

    const int tid  = threadIdx.x;
    const int lane = tid & 31;
    const int w    = tid >> 5;
    const int n0   = blockIdx.x * 64;
    const int wm   = (w & 3) * 16;   // 16 batch rows per wave
    const int wn   = (w >> 2) * 32;  // 32 H columns per wave

    // Stage this block's W_hh slice once (f32 -> f16)
#pragma unroll 4
    for (int i = 0; i < 32; ++i) {
        int id = tid + 256 * i;          // 8192 float4
        int r  = id >> 7;
        int c4 = (id & 127) << 2;
        float4 v = *(const float4*)(Whh + (size_t)(n0 + r) * HH + c4);
        v4h hv;
        hv[0] = (_Float16)v.x; hv[1] = (_Float16)v.y;
        hv[2] = (_Float16)v.z; hv[3] = (_Float16)v.w;
        *(v4h*)&Wsm[r * LDW + c4] = hv;
    }

    // ---- TDM descriptors (loop-invariant; one per h16 buffer parity) ----
    // Group0: count=1 | lds_addr | global_addr[56:0] | type=2
    const unsigned lds_hsm = (unsigned)(uintptr_t)(void*)Hsm;  // LDS aperture low bits = LDS offset
    const unsigned long long ga0 = (unsigned long long)(uintptr_t)h16;
    const unsigned long long ga1 = ga0 + (unsigned long long)BB * HH * 2;
    const v4u g0_buf0 = { 1u, lds_hsm, (unsigned)ga0, (unsigned)(ga0 >> 32) | 0x80000000u };
    const v4u g0_buf1 = { 1u, lds_hsm, (unsigned)ga1, (unsigned)(ga1 >> 32) | 0x80000000u };
    // Group1: data_size=2B(1<<16) | pad_enable(1<<20) | pad_interval=7(<<22: 256 DW = 1024B)
    //         | pad_amount=3(<<25: 4 DW = 16B pad -> LDW=520 halves)
    // tensor_dim0=512, tensor_dim1=64, tile_dim0=512, tile_dim1=64, tile_dim2=0,
    // tensor_dim0_stride=512
    const v8i g1 = { (int)0x07D10000u,
                     (int)(512u << 16),   // tensor_dim0[15:0] @ bits 63:48
                     (int)(64u  << 16),   // tensor_dim1[15:0] @ bits 95:80 (tdim0 hi = 0)
                     (int)(512u << 16),   // tile_dim0 @ bits 127:112
                     (int)64,             // tile_dim1 @ bits 143:128
                     (int)512,            // tensor_dim0_stride[31:0]
                     0, 0 };
    const v4i gz4 = { 0, 0, 0, 0 };         // groups 2/3: unused (2D tensor)
    const v8i gz8 = { 0, 0, 0, 0, 0, 0, 0, 0 };  // trailing group (6-arg form): unused

    for (int t = 0; t < SEQ; ++t) {
        // Stage h_{t-1}: TDM DMA f16 [64 x 512] global -> padded LDS tile.
        // Read parity (t+1)&1: t=0 reads buf1 (initialized with hx by init kernel).
        if (tid < 32) {
            __builtin_amdgcn_tensor_load_to_lds(((t & 1) == 0) ? g0_buf1 : g0_buf0,
                                                g1, gz4, gz4, gz8, 0);
            __builtin_amdgcn_s_wait_tensorcnt(0);
        }
        __syncthreads();

        v8f c0 = {}, c1 = {};
#pragma unroll
        for (int kt = 0; kt < HH / 32; ++kt) {
            v16h a  = frag_a(Hsm, wm,      LDW, kt * 32, lane);
            v16h b0 = frag_b(Wsm, wn,      LDW, kt * 32, lane);
            v16h b1 = frag_b(Wsm, wn + 16, LDW, kt * 32, lane);
            c0 = wmma_f16(a, b0, c0);
            c1 = wmma_f16(a, b1, c1);
        }

        // Epilogue: h_t = tanh(xp_t + acc); write f32 to out (in place over xp)
        // and f16 to the parity-(t&1) exchange buffer.
        const int    rsel = (lane & 16) ? 8 : 0;
        const int    cl   = lane & 15;
        const size_t base = (size_t)t * BB * HH;
        _Float16* hw = h16 + (size_t)(t & 1) * (BB * HH);
#pragma unroll
        for (int ni = 0; ni < 2; ++ni) {
            v8f cc = ni ? c1 : c0;
            const int col = n0 + wn + ni * 16 + cl;
#pragma unroll
            for (int v = 0; v < 8; ++v) {
                const int    brow = wm + rsel + v;
                const size_t idx  = base + (size_t)brow * HH + col;
                float val = tanhf(out[idx] + cc[v]);
                out[idx] = val;
                hw[(size_t)brow * HH + col] = (_Float16)val;
                if (t == SEQ - 1) {
                    out[(size_t)SEQ * BB * HH + (size_t)brow * HH + col] = val;
                }
            }
        }

        // Device-wide step barrier
        __threadfence();
        __syncthreads();
        if (tid == 0) {
            atomicAdd(&ctr[t], 1u);
            while (__hip_atomic_load(&ctr[t], __ATOMIC_RELAXED,
                                     __HIP_MEMORY_SCOPE_AGENT) < (unsigned)NBLK) {
                __builtin_amdgcn_s_sleep(2);
            }
        }
        __syncthreads();
    }
}

// ---- init: reset counters, seed f16 h buffer 1 with hx (graph-replay safe) -
__global__ void init_kernel(unsigned int* ctr, const float* __restrict__ hx,
                            _Float16* __restrict__ h16_buf1) {
    int i = blockIdx.x * blockDim.x + threadIdx.x;
    if (i < SEQ) ctr[i] = 0u;
    if (i < BB * HH) h16_buf1[i] = (_Float16)hx[i];
}

extern "C" void kernel_launch(void* const* d_in, const int* in_sizes, int n_in,
                              void* d_out, int out_size, void* d_ws, size_t ws_size,
                              hipStream_t stream) {
    const float* input = (const float*)d_in[0];  // [SEQ,BB,II]
    const float* hx    = (const float*)d_in[1];  // [1,BB,HH]
    const float* wih   = (const float*)d_in[2];  // [HH,II]
    const float* whh   = (const float*)d_in[3];  // [HH,HH]
    const float* bih   = (const float*)d_in[4];  // [HH]
    const float* bhh   = (const float*)d_in[5];  // [HH]
    float*        out  = (float*)d_out;

    unsigned int* ctr = (unsigned int*)d_ws;                       // 8 KB
    _Float16*     h16 = (_Float16*)((char*)d_ws + 65536);          // 2 x 64 KB

    init_kernel<<<128, 256, 0, stream>>>(ctr, hx, h16 + BB * HH);

    dim3 g1(MTOT / 128, HH / 256);
    xproj_kernel<<<g1, 256, 0, stream>>>(input, wih, bih, bhh, out);

    rnn_scan_kernel<<<NBLK, 256, 0, stream>>>(whh, out, ctr, h16);
}
